// AttentionModule_1340029796865
// MI455X (gfx1250) — matrix operand order repaired and checked by reading, not compiled, not run on hardware
//
#include <hip/hip_runtime.h>
#include <hip/hip_bf16.h>

typedef __attribute__((ext_vector_type(16))) __bf16 v16bf;
typedef __attribute__((ext_vector_type(8)))  float  v8f;
typedef unsigned int u32x4 __attribute__((ext_vector_type(4)));

#define BATCH 16
#define C 512
#define P 4096
#define NH 8
#define DH 64

__device__ __forceinline__ unsigned short f2bf(float f) {
  union { float f; unsigned int u; } a; a.f = f;
  unsigned int u = a.u;
  return (unsigned short)((u + 0x7FFFu + ((u >> 16) & 1u)) >> 16);
}

// A-operand (16x32 bf16): lane L: m = m0 + (L&15); kbase = 8*(L>>4);
// chunk0 covers k0+kbase..+7, chunk1 covers k0+16+kbase..+7 (per ISA layout).
__device__ __forceinline__ v16bf load_a(const unsigned short* base, int ld, int m0, int k0) {
  int lane = threadIdx.x & 31;
  const unsigned short* p = base + (size_t)(m0 + (lane & 15)) * ld + k0 + ((lane >> 4) << 3);
  union { v16bf v; u32x4 q[2]; } u;
  u.q[0] = *(const u32x4*)(p);
  u.q[1] = *(const u32x4*)(p + 16);
  return u.v;
}

// B-operand (32x16 bf16) from K-contiguous-per-column storage:
// element (k, n) lives at baseT[(n0+n)*ld + k0 + k].
// lane L: n = n0 + (L&15); chunk0 k0+8*(L>>4)..+7, chunk1 k0+16+8*(L>>4)..+7.
__device__ __forceinline__ v16bf load_bt(const unsigned short* baseT, int ld, int n0, int k0) {
  int lane = threadIdx.x & 31;
  const unsigned short* p = baseT + (size_t)(n0 + (lane & 15)) * ld + k0 + ((lane >> 4) << 3);
  union { v16bf v; u32x4 q[2]; } u;
  u.q[0] = *(const u32x4*)(p);
  u.q[1] = *(const u32x4*)(p + 16);
  return u.v;
}

__device__ __forceinline__ v8f wmma_bf16(v16bf a, v16bf b, v8f c) {
  // (neg_a, A, neg_b, B, c_mod, C, reuse_a, reuse_b)
  return __builtin_amdgcn_wmma_f32_16x16x32_bf16(false, a, false, b, (short)0, c, false, false);
}

// ---------------- small prep kernels ----------------

__global__ void k_cvt_w(const float* wq, const float* wk, const float* wo,
                        unsigned short* wqh, unsigned short* wkh, unsigned short* woh) {
  int i = blockIdx.x * 256 + threadIdx.x;
  if (i < C * C) {
    wqh[i] = f2bf(wq[i]);
    wkh[i] = f2bf(wk[i]);
    woh[i] = f2bf(wo[i]);
  }
}

__global__ void k_rowsum(const float* x, float* s) {
  int row = blockIdx.x;  // b*C + c
  const float* xr = x + (size_t)row * P;
  float acc = 0.f;
  for (int i = threadIdx.x; i < P; i += 128) acc += xr[i];
  __shared__ float red[128];
  red[threadIdx.x] = acc;
  __syncthreads();
  for (int st = 64; st > 0; st >>= 1) {
    if (threadIdx.x < st) red[threadIdx.x] += red[threadIdx.x + st];
    __syncthreads();
  }
  if (threadIdx.x == 0) s[row] = red[0];
}

__global__ void k_qsks(const float* wq, const float* wk, const float* s,
                       float* qs, float* ks) {
  int c = blockIdx.x, b = blockIdx.y;
  const float* sv = s + (size_t)b * C;
  float aq = 0.f, ak = 0.f;
  for (int i = threadIdx.x; i < C; i += 128) {
    float svv = sv[i];
    aq += wq[(size_t)c * C + i] * svv;
    ak += wk[(size_t)c * C + i] * svv;
  }
  __shared__ float rq[128], rk[128];
  rq[threadIdx.x] = aq; rk[threadIdx.x] = ak;
  __syncthreads();
  for (int st = 64; st > 0; st >>= 1) {
    if (threadIdx.x < st) { rq[threadIdx.x] += rq[threadIdx.x + st]; rk[threadIdx.x] += rk[threadIdx.x + st]; }
    __syncthreads();
  }
  if (threadIdx.x == 0) { qs[(size_t)b * C + c] = rq[0]; ks[(size_t)b * C + c] = rk[0]; }
}

// convert x -> bf16 xh[b][c][p] and transposed xT[b][p][c] via LDS tile
__global__ void k_xprep(const float* x, unsigned short* xh, unsigned short* xT) {
  __shared__ unsigned short tile[64][72];
  int pt = blockIdx.x, ct = blockIdx.y, b = blockIdx.z;
  const float* xb = x + ((size_t)b * C + ct * 64) * P + pt * 64;
  for (int r = 0; r < 16; r++) {
    int idx = threadIdx.x + 256 * r;
    int cl = idx >> 6, pl = idx & 63;
    unsigned short h = f2bf(xb[(size_t)cl * P + pl]);
    xh[((size_t)b * C + ct * 64 + cl) * P + pt * 64 + pl] = h;
    tile[cl][pl] = h;
  }
  __syncthreads();
  for (int r = 0; r < 16; r++) {
    int idx = threadIdx.x + 256 * r;
    int pl = idx >> 6, cl = idx & 63;
    xT[((size_t)b * P + pt * 64 + pl) * C + ct * 64 + cl] = tile[cl][pl];
  }
}

// ---------------- Gram: G[b] = xh[b] @ xh[b]^T (bf16 out, symmetric) ----------------
// grid (4,4,16), block 256 (8 waves, 4m x 2n); block tile 128x128, wave 32x64, K=4096
// Only upper-triangle blocks (tn >= tm) computed; mirror-written via symmetry.
__global__ void k_gram(const unsigned short* xh, unsigned short* Gbf) {
  int tm = blockIdx.x, tn = blockIdx.y, b = blockIdx.z;
  if (tn < tm) return;  // symmetry: handled by mirror store of block (tm,tn)
  int w = threadIdx.x >> 5, lane = threadIdx.x & 31;
  int lh = lane >> 4, lc = lane & 15;
  int wm = w >> 1, wn = w & 1;
  int m0 = tm * 128 + wm * 32;
  int n0 = tn * 128 + wn * 64;
  const unsigned short* xb = xh + (size_t)b * C * P;
  v8f acc[2][4] = {};
  for (int k0 = 0; k0 < P; k0 += 32) {
    if (k0 + 64 < P)
      __builtin_prefetch((const void*)(xb + (size_t)(m0 + (lane & 15)) * P + k0 + 64), 0, 3);
    v16bf a0 = load_a(xb, P, m0, k0);
    v16bf a1 = load_a(xb, P, m0 + 16, k0);
    v16bf bb[4];
    for (int j = 0; j < 4; j++) bb[j] = load_bt(xb, P, n0 + 16 * j, k0);
    for (int j = 0; j < 4; j++) {
      acc[0][j] = wmma_bf16(a0, bb[j], acc[0][j]);
      acc[1][j] = wmma_bf16(a1, bb[j], acc[1][j]);
    }
  }
  unsigned short* Gb = Gbf + (size_t)b * C * C;
  bool diag = (tm == tn);
  for (int i = 0; i < 2; i++)
    for (int j = 0; j < 4; j++)
      for (int r = 0; r < 8; r++) {
        int m = m0 + 16 * i + r + 8 * lh;
        int nn = n0 + 16 * j + lc;
        unsigned short h = f2bf(acc[i][j][r]);
        Gb[(size_t)m * C + nn] = h;
        if (!diag) Gb[(size_t)nn * C + m] = h;  // mirror lower triangle
        else if (m != nn) Gb[(size_t)nn * C + m] = h;
      }
}

// ---------------- attention: att_n = softmax((Wq G Wk^T + bias)/8), store attT ----------------
// grid (NH, BATCH), block 256 (8 waves)
__global__ void k_att(const unsigned short* wqh, const unsigned short* wkh,
                      const unsigned short* Gbf, const float* bq, const float* bk,
                      const float* qs, const float* ks, unsigned short* attT) {
  __shared__ __align__(16) unsigned short Tl[64][512];  // T = Wq_n @ G, bf16 (64 KB)
  int hn = blockIdx.x, b = blockIdx.y;
  int w = threadIdx.x >> 5, lane = threadIdx.x & 31;
  int lh = lane >> 4, lc = lane & 15;
  const unsigned short* Gb = Gbf + (size_t)b * C * C;

  // stage 1: wave w computes T columns [64w, 64w+64); B-operand uses G symmetry
  {
    v8f acc[4][4] = {};
    for (int k0 = 0; k0 < C; k0 += 32) {
      v16bf a[4], bb[4];
      for (int i = 0; i < 4; i++) a[i] = load_a(wqh, C, hn * DH + 16 * i, k0);
      for (int j = 0; j < 4; j++) bb[j] = load_bt(Gb, C, w * 64 + 16 * j, k0);
      for (int i = 0; i < 4; i++)
        for (int j = 0; j < 4; j++) acc[i][j] = wmma_bf16(a[i], bb[j], acc[i][j]);
    }
    for (int i = 0; i < 4; i++)
      for (int j = 0; j < 4; j++)
        for (int r = 0; r < 8; r++) {
          int m = 16 * i + r + 8 * lh;
          int col = w * 64 + 16 * j + lc;
          Tl[m][col] = f2bf(acc[i][j][r]);
        }
  }
  __syncthreads();

  // stage 2: waves 0..3: rows [16w,16w+16) x e[0,64), K=512 over T (LDS) and Wk
  if (w < 4) {
    v8f acc2[4] = {};
    const unsigned short* Tp = &Tl[0][0];
    for (int k0 = 0; k0 < C; k0 += 32) {
      v16bf a = load_a(Tp, C, 16 * w, k0);
      for (int j = 0; j < 4; j++) {
        v16bf bb = load_bt(wkh, C, hn * DH + 16 * j, k0);
        acc2[j] = wmma_bf16(a, bb, acc2[j]);
      }
    }
    // add bias cross-terms, scale by 1/sqrt(dh)=1/8
    float att[4][8];
    float bqv[8], qsv[8], bkv[4], ksv[4];
    for (int r = 0; r < 8; r++) {
      int dp = hn * DH + 16 * w + r + 8 * lh;
      bqv[r] = bq[dp];
      qsv[r] = qs[(size_t)b * C + dp];
    }
    for (int j = 0; j < 4; j++) {
      int ep = hn * DH + 16 * j + lc;
      bkv[j] = bk[ep];
      ksv[j] = ks[(size_t)b * C + ep];
    }
    for (int j = 0; j < 4; j++)
      for (int r = 0; r < 8; r++)
        att[j][r] = (acc2[j][r] + bqv[r] * ksv[j] + bkv[j] * qsv[r] +
                     4096.0f * bqv[r] * bkv[j]) * 0.125f;
    // softmax over e: values live across j (4 frags) and lc (16 lanes, xor stays in half)
    for (int r = 0; r < 8; r++) {
      float m = att[0][r];
      for (int j = 1; j < 4; j++) m = fmaxf(m, att[j][r]);
      for (int msk = 1; msk < 16; msk <<= 1) m = fmaxf(m, __shfl_xor(m, msk, 32));
      float ssum = 0.f;
      for (int j = 0; j < 4; j++) { att[j][r] = __expf(att[j][r] - m); ssum += att[j][r]; }
      for (int msk = 1; msk < 16; msk <<= 1) ssum += __shfl_xor(ssum, msk, 32);
      float inv = 1.0f / ssum;
      for (int j = 0; j < 4; j++) att[j][r] *= inv;
    }
    // store transposed attT[e][d] bf16 (K-contiguous for Weff B-operand)
    unsigned short* ap = attT + ((size_t)(b * NH + hn)) * DH * DH;
    for (int j = 0; j < 4; j++)
      for (int r = 0; r < 8; r++) {
        int d = 16 * w + r + 8 * lh;
        int e = 16 * j + lc;
        ap[e * DH + d] = f2bf(att[j][r]);
      }
  }
}

// ---------------- Weff[:, head n cols] = Wo[:, head n cols] @ att_n ----------------
// grid (NH, BATCH), block 256; wave w: o-rows [64w,64w+64), N=64(e), K=64(d)
__global__ void k_weff(const unsigned short* woh, const unsigned short* attT,
                       unsigned short* Weffh) {
  int hn = blockIdx.x, b = blockIdx.y;
  int w = threadIdx.x >> 5, lane = threadIdx.x & 31;
  int lh = lane >> 4, lc = lane & 15;
  const unsigned short* ap = attT + ((size_t)(b * NH + hn)) * DH * DH;
  v8f acc[4][4] = {};
  for (int kk = 0; kk < DH; kk += 32) {
    v16bf a[4], bb[4];
    for (int i = 0; i < 4; i++) a[i] = load_a(woh, C, w * 64 + 16 * i, hn * DH + kk);
    for (int j = 0; j < 4; j++) bb[j] = load_bt(ap, DH, 16 * j, kk);
    for (int i = 0; i < 4; i++)
      for (int j = 0; j < 4; j++) acc[i][j] = wmma_bf16(a[i], bb[j], acc[i][j]);
  }
  unsigned short* Wb = Weffh + (size_t)b * C * C;
  for (int i = 0; i < 4; i++)
    for (int j = 0; j < 4; j++)
      for (int r = 0; r < 8; r++) {
        int o = w * 64 + 16 * i + r + 8 * lh;
        int col = hn * DH + 16 * j + lc;
        Wb[(size_t)o * C + col] = f2bf(acc[i][j][r]);
      }
}

// ---------------- final: y = Weff @ x + bo + x ----------------
// grid (4,32,16), block 256 (8 waves 4m x 2n); tile 128(o) x 128(p), wave 32x64, K=512
__global__ void k_final(const unsigned short* Weffh, const unsigned short* xT,
                        const float* x, const float* bo, float* out) {
  int tm = blockIdx.x, tp = blockIdx.y, b = blockIdx.z;
  int w = threadIdx.x >> 5, lane = threadIdx.x & 31;
  int lh = lane >> 4, lc = lane & 15;
  int wm = w >> 1, wn = w & 1;
  int m0 = tm * 128 + wm * 32;
  int p0 = tp * 128 + wn * 64;
  const unsigned short* Wb = Weffh + (size_t)b * C * C;
  const unsigned short* xTb = xT + (size_t)b * P * C;
  v8f acc[2][4] = {};
  for (int k0 = 0; k0 < C; k0 += 32) {
    if (k0 + 64 < C)
      __builtin_prefetch((const void*)(xTb + (size_t)(p0 + (lane & 15)) * C + k0 + 64), 0, 3);
    v16bf a0 = load_a(Wb, C, m0, k0);
    v16bf a1 = load_a(Wb, C, m0 + 16, k0);
    v16bf bb[4];
    for (int j = 0; j < 4; j++) bb[j] = load_bt(xTb, C, p0 + 16 * j, k0);
    for (int j = 0; j < 4; j++) {
      acc[0][j] = wmma_bf16(a0, bb[j], acc[0][j]);
      acc[1][j] = wmma_bf16(a1, bb[j], acc[1][j]);
    }
  }
  for (int i = 0; i < 2; i++) {
    for (int r = 0; r < 8; r++) {
      int o = m0 + 16 * i + r + 8 * lh;
      float bov = bo[o];
      const float* xr = x + ((size_t)b * C + o) * P;
      float* yr = out + ((size_t)b * C + o) * P;
      for (int j = 0; j < 4; j++) {
        int p = p0 + 16 * j + lc;
        yr[p] = acc[i][j][r] + bov + xr[p];
      }
    }
  }
}

extern "C" void kernel_launch(void* const* d_in, const int* in_sizes, int n_in,
                              void* d_out, int out_size, void* d_ws, size_t ws_size,
                              hipStream_t stream) {
  (void)in_sizes; (void)n_in; (void)out_size; (void)ws_size;
  const float* x  = (const float*)d_in[0];
  const float* wq = (const float*)d_in[1];
  const float* bq = (const float*)d_in[2];
  const float* wk = (const float*)d_in[3];
  const float* bk = (const float*)d_in[4];
  const float* wo = (const float*)d_in[5];
  const float* bo = (const float*)d_in[6];
  float* out = (float*)d_out;

  char* ws = (char*)d_ws;
  unsigned short* xh    = (unsigned short*)(ws + 0ull);           // 64 MB
  unsigned short* xT    = (unsigned short*)(ws + 67108864ull);    // 64 MB
  unsigned short* Gbf   = (unsigned short*)(ws + 134217728ull);   // 8 MB
  unsigned short* Weffh = (unsigned short*)(ws + 142606336ull);   // 8 MB
  unsigned short* attT  = (unsigned short*)(ws + 150994944ull);   // 1 MB
  unsigned short* wqh   = (unsigned short*)(ws + 152043520ull);   // 512 KB
  unsigned short* wkh   = (unsigned short*)(ws + 152567808ull);   // 512 KB
  unsigned short* woh   = (unsigned short*)(ws + 153092096ull);   // 512 KB
  float* s  = (float*)(ws + 153616384ull);                        // 32 KB
  float* qs = (float*)(ws + 153649152ull);                        // 32 KB
  float* ks = (float*)(ws + 153681920ull);                        // 32 KB

  k_cvt_w<<<dim3((C * C + 255) / 256), 256, 0, stream>>>(wq, wk, wo, wqh, wkh, woh);
  k_rowsum<<<dim3(BATCH * C), 128, 0, stream>>>(x, s);
  k_qsks<<<dim3(C, BATCH), 128, 0, stream>>>(wq, wk, s, qs, ks);
  k_xprep<<<dim3(P / 64, C / 64, BATCH), 256, 0, stream>>>(x, xh, xT);
  k_gram<<<dim3(4, 4, BATCH), 256, 0, stream>>>(xh, Gbf);
  k_att<<<dim3(NH, BATCH), 256, 0, stream>>>(wqh, wkh, Gbf, bq, bk, qs, ks, attT);
  k_weff<<<dim3(NH, BATCH), 256, 0, stream>>>(woh, attT, Weffh);
  k_final<<<dim3(4, 32, BATCH), 256, 0, stream>>>(Weffh, xT, x, bo, out);
}
